// BraskModel_77429670412702
// MI455X (gfx1250) — matrix operand order repaired and checked by reading, not compiled, not run on hardware
//
#include <hip/hip_runtime.h>
#include <hip/hip_bf16.h>

// Shapes from the reference
// B=4, R=8, S=8, L=256, H=768, A=256, TD=100
// Flagship GEMM: wx = X @ Wx -> M=B*L=1024, K=H=768, N=A=256 (x2 dirs fused: N=512)

typedef __attribute__((ext_vector_type(16))) __bf16 v16bf;
typedef __attribute__((ext_vector_type(8)))  float v8f;

__device__ inline unsigned short f2bf(float x) {
  unsigned int u = __float_as_uint(x);
  unsigned int r = (u + 0x7FFFu + ((u >> 16) & 1u)) >> 16;  // round-to-nearest-even
  return (unsigned short)r;
}

// ---- prep: X -> bf16 row-major; Wx (768x256) -> bf16 fragment-major ----
// W packing: for K-chunk c=k>>5, lane-group g=(k>>4)&1, v=(k>>1)&7, p=k&1, col n:
//   dword[(((c*2+g)<<8) + n)*8 + v] = pair (k even in [15:0], k odd in [31:16])
// so each lane's 8 B-fragment dwords are contiguous -> 2x b128 loads.
__global__ void k_convert(const float* __restrict__ X,
                          const float* __restrict__ Wf, const float* __restrict__ Wb,
                          unsigned short* __restrict__ Xbf,
                          unsigned short* __restrict__ Wfpk,
                          unsigned short* __restrict__ Wbpk) {
  int i = blockIdx.x * 256 + threadIdx.x;
  if (i < 786432) Xbf[i] = f2bf(X[i]);
  if (i < 196608) {                 // 768*256, row-major (k,n)
    int k = i >> 8, n = i & 255;
    int c = k >> 5, g = (k >> 4) & 1, v = (k >> 1) & 7, p = k & 1;
    int dstd = ((((c << 1) + g) << 8) + n) * 8 + v;
    Wfpk[dstd * 2 + p] = f2bf(Wf[i]);
    Wbpk[dstd * 2 + p] = f2bf(Wb[i]);
  }
}

// ---- WMMA GEMM: [1024 x 512] ; wave tile = 16 rows x 64 cols (4 accumulators) ----
__global__ void k_gemm_wmma(const unsigned short* __restrict__ Xbf,
                            const unsigned int* __restrict__ Wpkf,
                            const unsigned int* __restrict__ Wpkb,
                            float* __restrict__ wx_f, float* __restrict__ wx_b) {
  const int K = 768, N = 256;
  int wave = (blockIdx.x * blockDim.x + threadIdx.x) >> 5;  // 512 waves
  int lane = threadIdx.x & 31;
  int tM = wave >> 3;             // 64 M-tiles
  int nG = wave & 7;              // 8 N-groups of 64 over fused N=512
  int m0 = tM << 4;
  int g  = lane >> 4;
  int mr = lane & 15;

  bool bwd = (nG >= 4);
  const unsigned int* Wd = bwd ? Wpkb : Wpkf;
  float* out = bwd ? wx_b : wx_f;
  int n0 = (nG & 3) << 6;         // 0,64,128,192 within this direction

  const uint4* Arow = (const uint4*)(Xbf + (size_t)(m0 + mr) * K);
  const uint4* Bq   = (const uint4*)Wd;

  v8f z = {0.f, 0.f, 0.f, 0.f, 0.f, 0.f, 0.f, 0.f};
  v8f acc0 = z, acc1 = z, acc2 = z, acc3 = z;
  union U { v16bf v; uint4 q[2]; } fa, fb0, fb1, fb2, fb3;

  for (int c = 0; c < 24; ++c) {
    // A 16x32 bf16 fragment: dwords (k0+16g)/2 + {0..7} -> 2 aligned uint4
    fa.q[0] = Arow[(c << 2) + (g << 1)];
    fa.q[1] = Arow[(c << 2) + (g << 1) + 1];
    // B fragments: 8 contiguous dwords per lane = 2 uint4, column = n0 + 16*t + mr
    int cb = (((c << 1) + g) << 8);
    int b0 = (cb + n0 + mr) << 1;
    fb0.q[0] = Bq[b0];      fb0.q[1] = Bq[b0 + 1];
    fb1.q[0] = Bq[b0 + 32]; fb1.q[1] = Bq[b0 + 33];   // +16 cols = +32 uint4
    fb2.q[0] = Bq[b0 + 64]; fb2.q[1] = Bq[b0 + 65];
    fb3.q[0] = Bq[b0 + 96]; fb3.q[1] = Bq[b0 + 97];
    acc0 = __builtin_amdgcn_wmma_f32_16x16x32_bf16(false, fa.v, false, fb0.v, (short)0, acc0, false, false);
    acc1 = __builtin_amdgcn_wmma_f32_16x16x32_bf16(false, fa.v, false, fb1.v, (short)0, acc1, false, false);
    acc2 = __builtin_amdgcn_wmma_f32_16x16x32_bf16(false, fa.v, false, fb2.v, (short)0, acc2, false, false);
    acc3 = __builtin_amdgcn_wmma_f32_16x16x32_bf16(false, fa.v, false, fb3.v, (short)0, acc3, false, false);
  }
  // C/D layout: VGPR v -> row m0 + v + 8*g, col = lane&15 within each 16-wide tile
#pragma unroll
  for (int v = 0; v < 8; ++v) {
    size_t row = (size_t)(m0 + v + (g << 3)) * N;
    out[row + n0 + mr]      = acc0[v];
    out[row + n0 + 16 + mr] = acc1[v];
    out[row + n0 + 32 + mr] = acc2[v];
    out[row + n0 + 48 + mr] = acc3[v];
  }
}

// ---- generic tiny GEMM: Y[M,N] = A[M,K] @ W[K,N] + bias[N] ----
__global__ void k_mat_small(const float* __restrict__ Am, const float* __restrict__ W,
                            const float* __restrict__ bias, float* __restrict__ Y,
                            int M, int K, int N) {
  int i = blockIdx.x * blockDim.x + threadIdx.x;
  if (i >= M * N) return;
  int m = i / N, n = i % N;
  float acc = bias ? bias[n] : 0.f;
  const float* row = Am + (size_t)m * K;
  for (int k = 0; k < K; ++k) acc += row[k] * W[(size_t)k * N + n];
  Y[i] = acc;
}

// ---- batched row-dots: Y[i] = Mtx[i,:]·vec + addv[i] + *bias, one job per blockIdx.y ----
struct RowDotJob {
  const float* mtx; const float* vec; const float* addv; const float* bias;
  float* y; int M; int K;
};
struct RowDotBatch { RowDotJob j[20]; };

__global__ void k_rowdot_batch(RowDotBatch batch) {
  RowDotJob jb = batch.j[blockIdx.y];
  int i = blockIdx.x * blockDim.x + threadIdx.x;
  if (i >= jb.M) return;
  const float* row = jb.mtx + (size_t)i * jb.K;
  float acc = 0.f;
  for (int k = 0; k < jb.K; ++k) acc += row[k] * jb.vec[k];
  if (jb.addv) acc += jb.addv[i];
  if (jb.bias) acc += jb.bias[0];
  jb.y[i] = acc;
}

// ---- span endpoint gather: sk_embs[b,r,s,h] = skm * 0.5*(X[b,sk0,h]+X[b,sk1,h]) ----
__global__ void k_skembs(const float* __restrict__ X, const int* __restrict__ sk,
                         const float* __restrict__ skm, float* __restrict__ outp) {
  int i = blockIdx.x * blockDim.x + threadIdx.x;
  if (i >= 256 * 768) return;        // (B*R*S) * H
  int h = i % 768, brs = i / 768, b = brs >> 6;
  int i0 = sk[brs * 2 + 0], i1 = sk[brs * 2 + 1];
  float e = 0.5f * (X[(size_t)(b * 256 + i0) * 768 + h] + X[(size_t)(b * 256 + i1) * 768 + h]);
  outp[i] = skm[brs] * e;
}

// ---- attention energies: e[b,r,l] = V · tanh(wx+bx+wr+wg) + bv ----
__global__ void k_e(const float* __restrict__ wx, const float* __restrict__ bx,
                    const float* __restrict__ wr, const float* __restrict__ wg,
                    const float* __restrict__ V, const float* __restrict__ bv,
                    float* __restrict__ e) {
  int i = blockIdx.x * blockDim.x + threadIdx.x;
  if (i >= 4 * 8 * 256) return;
  int l = i & 255, br = i >> 8, r = br & 7, b = br >> 3;
  const float* px = wx + (size_t)(b * 256 + l) * 256;
  const float* pr = wr + r * 256;
  const float* pg = wg + b * 256;
  float acc = 0.f;
  for (int a = 0; a < 256; ++a)
    acc += V[a] * tanhf(px[a] + bx[a] + pr[a] + pg[a]);
  e[i] = acc + bv[0];
}

// ---- softmax over L=256, one block per (b,r) row, in place ----
__global__ void k_softmax(float* __restrict__ e) {
  __shared__ float red[256];
  int row = blockIdx.x, t = threadIdx.x;
  float v = e[row * 256 + t];
  red[t] = v; __syncthreads();
  for (int s = 128; s > 0; s >>= 1) { if (t < s) red[t] = fmaxf(red[t], red[t + s]); __syncthreads(); }
  float mx = red[0]; __syncthreads();
  float ex = expf(v - mx);
  red[t] = ex; __syncthreads();
  for (int s = 128; s > 0; s >>= 1) { if (t < s) red[t] += red[t + s]; __syncthreads(); }
  e[row * 256 + t] = ex / red[0];
}

// ---- context: c[b,r,h] = sum_l a[b,r,l] * X[b,l,h] ----
__global__ void k_ctx(const float* __restrict__ a, const float* __restrict__ X,
                      float* __restrict__ c) {
  int i = blockIdx.x * blockDim.x + threadIdx.x;
  if (i >= 32 * 768) return;
  int h = i % 768, br = i / 768, b = br >> 3;
  float acc = 0.f;
  for (int l = 0; l < 256; ++l)
    acc += a[br * 256 + l] * X[(size_t)(b * 256 + l) * 768 + h];
  c[i] = acc;
}

// ---- Q[d,brs] = skm * (sk_embs[brs,:]·vecA[d] + scalS[d]) ----
__global__ void k_Q(const float* __restrict__ skemb, const float* __restrict__ skm,
                    const float* __restrict__ vecA, const float* __restrict__ scalS,
                    float* __restrict__ Q) {
  int i = blockIdx.x * blockDim.x + threadIdx.x;
  if (i >= 4 * 256) return;
  int d = i >> 8, brs = i & 255;
  const float* row = skemb + (size_t)brs * 768;
  const float* v = vecA + d * 768;
  float acc = 0.f;
  for (int h = 0; h < 768; ++h) acc += row[h] * v[h];
  Q[i] = skm[brs] * (acc + scalS[d]);
}

// ---- final broadcast add: out[d][b,r,s,l] = Q + G + P + bias_d ----
__global__ void k_final(const float* __restrict__ P, const float* __restrict__ Q,
                        const float* __restrict__ G,
                        const float* __restrict__ b0, const float* __restrict__ b1,
                        const float* __restrict__ b2, const float* __restrict__ b3,
                        float* __restrict__ out) {
  int i = blockIdx.x * blockDim.x + threadIdx.x;
  if (i >= 4 * 65536) return;
  int d = i >> 16, brsl = i & 65535;
  int l = brsl & 255, brs = brsl >> 8, br = brs >> 3, b = brs >> 6;
  float bias = (d == 0) ? b0[0] : (d == 1) ? b1[0] : (d == 2) ? b2[0] : b3[0];
  out[i] = Q[d * 256 + brs] + G[d * 32 + br] + P[d * 1024 + b * 256 + l] + bias;
}

extern "C" void kernel_launch(void* const* d_in, const int* in_sizes, int n_in,
                              void* d_out, int out_size, void* d_ws, size_t ws_size,
                              hipStream_t stream) {
  (void)in_sizes; (void)n_in; (void)out_size; (void)ws_size;
  const float* X      = (const float*)d_in[0];
  const float* X_mean = (const float*)d_in[1];
  // d_in[2] = mask (unused by reference outputs)
  const int*   sk      = (const int*)d_in[3];
  const float* sk_mask = (const float*)d_in[4];
  const float* sem     = (const float*)d_in[5];
  const float* tre     = (const float*)d_in[6];
  const float *fh_ws=(const float*)d_in[7],  *fh_bs=(const float*)d_in[8],
              *fh_we=(const float*)d_in[9],  *fh_be=(const float*)d_in[10];
  const float *bt_ws=(const float*)d_in[11], *bt_bs=(const float*)d_in[12],
              *bt_we=(const float*)d_in[13], *bt_be=(const float*)d_in[14];
  const float *ft_ws=(const float*)d_in[15], *ft_bs=(const float*)d_in[16],
              *ft_we=(const float*)d_in[17], *ft_be=(const float*)d_in[18];
  const float *bh_ws=(const float*)d_in[19], *bh_bs=(const float*)d_in[20],
              *bh_we=(const float*)d_in[21], *bh_be=(const float*)d_in[22];
  const float *fra_Wx=(const float*)d_in[23], *fra_bx=(const float*)d_in[24],
              *fra_Wr=(const float*)d_in[25], *fra_br=(const float*)d_in[26],
              *fra_Wg=(const float*)d_in[27], *fra_bg=(const float*)d_in[28],
              *fra_V =(const float*)d_in[29], *fra_bv=(const float*)d_in[30];
  const float *bra_Wx=(const float*)d_in[31], *bra_bx=(const float*)d_in[32],
              *bra_Wr=(const float*)d_in[33], *bra_br=(const float*)d_in[34],
              *bra_Wg=(const float*)d_in[35], *bra_bg=(const float*)d_in[36],
              *bra_V =(const float*)d_in[37], *bra_bv=(const float*)d_in[38];
  const float *ffe_Ws=(const float*)d_in[39], *ffe_bs=(const float*)d_in[40],
              *ffe_Wx=(const float*)d_in[41], *ffe_bx=(const float*)d_in[42];
  const float *bfe_Ws=(const float*)d_in[43], *bfe_bs=(const float*)d_in[44],
              *bfe_Wx=(const float*)d_in[45], *bfe_bx=(const float*)d_in[46];

  float* out = (float*)d_out;

  // ---- workspace carve-out (256B aligned chunks) ----
  char* w = (char*)d_ws;
  size_t off = 0;
  auto take = [&](size_t bytes) -> char* {
    char* p = w + off;
    off += bytes;
    off = (off + 255) & ~(size_t)255;
    return p;
  };
  unsigned short* Xbf  = (unsigned short*)take(786432 * 2);
  unsigned short* Wfpk = (unsigned short*)take(196608 * 2);
  unsigned short* Wbpk = (unsigned short*)take(196608 * 2);
  float* wx_f  = (float*)take(262144 * 4);
  float* wx_b  = (float*)take(262144 * 4);
  float* wr_f  = (float*)take(2048 * 4);
  float* wr_b  = (float*)take(2048 * 4);
  float* wg_f  = (float*)take(1024 * 4);
  float* wg_b  = (float*)take(1024 * 4);
  float* att_f = (float*)take(8192 * 4);
  float* att_b = (float*)take(8192 * 4);
  float* c_f   = (float*)take(24576 * 4);
  float* c_b   = (float*)take(24576 * 4);
  float* skemb = (float*)take(196608 * 4);
  float* vecP  = (float*)take(4 * 768 * 4);
  float* vecA  = (float*)take(4 * 768 * 4);
  float* scalX = (float*)take(4 * 4);
  float* scalS = (float*)take(4 * 4);
  float* Pm    = (float*)take(4 * 1024 * 4);
  float* Qm    = (float*)take(4 * 256 * 4);
  float* Gm    = (float*)take(4 * 32 * 4);

  // ---- prep: bf16 conversion + fragment-major W packing ----
  k_convert<<<(786432 + 255) / 256, 256, 0, stream>>>(X, fra_Wx, bra_Wx, Xbf, Wfpk, Wbpk);

  // ---- small attention projections ----
  k_mat_small<<<8, 256, 0, stream>>>(sem, fra_Wr, fra_br, wr_f, 8, 768, 256);
  k_mat_small<<<8, 256, 0, stream>>>(tre, bra_Wr, bra_br, wr_b, 8, 100, 256);
  k_mat_small<<<4, 256, 0, stream>>>(X_mean, fra_Wg, fra_bg, wg_f, 4, 768, 256);
  k_mat_small<<<4, 256, 0, stream>>>(X_mean, bra_Wg, bra_bg, wg_b, 4, 768, 256);

  // ---- batch 1: first-stage entity logits + reduced vectors + scalars (20 jobs) ----
  RowDotBatch b1 = {};
  // entity logits over X straight into d_out
  b1.j[0]  = {X, fh_ws, nullptr, fh_bs, out + 0,    1024, 768};
  b1.j[1]  = {X, fh_we, nullptr, fh_be, out + 1024, 1024, 768};
  b1.j[2]  = {X, bt_ws, nullptr, bt_bs, out + 2048, 1024, 768};
  b1.j[3]  = {X, bt_we, nullptr, bt_be, out + 3072, 1024, 768};
  // vecP[d] = Wx@ws + ws ; vecA[d] = Ws@ws
  b1.j[4]  = {ffe_Wx, ft_ws, ft_ws, nullptr, vecP + 0,    768, 768};
  b1.j[5]  = {ffe_Wx, ft_we, ft_we, nullptr, vecP + 768,  768, 768};
  b1.j[6]  = {bfe_Wx, bh_ws, bh_ws, nullptr, vecP + 1536, 768, 768};
  b1.j[7]  = {bfe_Wx, bh_we, bh_we, nullptr, vecP + 2304, 768, 768};
  b1.j[8]  = {ffe_Ws, ft_ws, nullptr, nullptr, vecA + 0,    768, 768};
  b1.j[9]  = {ffe_Ws, ft_we, nullptr, nullptr, vecA + 768,  768, 768};
  b1.j[10] = {bfe_Ws, bh_ws, nullptr, nullptr, vecA + 1536, 768, 768};
  b1.j[11] = {bfe_Ws, bh_we, nullptr, nullptr, vecA + 2304, 768, 768};
  // scalars bx·ws, bs·ws
  b1.j[12] = {ffe_bx, ft_ws, nullptr, nullptr, scalX + 0, 1, 768};
  b1.j[13] = {ffe_bx, ft_we, nullptr, nullptr, scalX + 1, 1, 768};
  b1.j[14] = {bfe_bx, bh_ws, nullptr, nullptr, scalX + 2, 1, 768};
  b1.j[15] = {bfe_bx, bh_we, nullptr, nullptr, scalX + 3, 1, 768};
  b1.j[16] = {ffe_bs, ft_ws, nullptr, nullptr, scalS + 0, 1, 768};
  b1.j[17] = {ffe_bs, ft_we, nullptr, nullptr, scalS + 1, 1, 768};
  b1.j[18] = {bfe_bs, bh_ws, nullptr, nullptr, scalS + 2, 1, 768};
  b1.j[19] = {bfe_bs, bh_we, nullptr, nullptr, scalS + 3, 1, 768};
  k_rowdot_batch<<<dim3(4, 20), 256, 0, stream>>>(b1);

  // ---- batch 2: P[d, b*L+l] = X·vecP[d] + scalX[d] (depends on batch 1) ----
  RowDotBatch b2 = {};
  b2.j[0] = {X, vecP + 0,    nullptr, scalX + 0, Pm + 0,    1024, 768};
  b2.j[1] = {X, vecP + 768,  nullptr, scalX + 1, Pm + 1024, 1024, 768};
  b2.j[2] = {X, vecP + 1536, nullptr, scalX + 2, Pm + 2048, 1024, 768};
  b2.j[3] = {X, vecP + 2304, nullptr, scalX + 3, Pm + 3072, 1024, 768};
  k_rowdot_batch<<<dim3(4, 4), 256, 0, stream>>>(b2);

  // ---- span embeddings + Q term ----
  k_skembs<<<(196608 + 255) / 256, 256, 0, stream>>>(X, sk, sk_mask, skemb);
  k_Q<<<4, 256, 0, stream>>>(skemb, sk_mask, vecA, scalS, Qm);

  // ---- flagship WMMA GEMM (512 waves, 16x64 tile/wave, 4 WMMA per K-step) ----
  k_gemm_wmma<<<64, 256, 0, stream>>>(Xbf, (const unsigned int*)Wfpk,
                                      (const unsigned int*)Wbpk, wx_f, wx_b);

  // ---- attention: energies, softmax, context ----
  k_e<<<32, 256, 0, stream>>>(wx_f, fra_bx, wr_f, wg_f, fra_V, fra_bv, att_f);
  k_e<<<32, 256, 0, stream>>>(wx_b, bra_bx, wr_b, wg_b, bra_V, bra_bv, att_b);
  k_softmax<<<32, 256, 0, stream>>>(att_f);
  k_softmax<<<32, 256, 0, stream>>>(att_b);
  k_ctx<<<96, 256, 0, stream>>>(att_f, X, c_f);
  k_ctx<<<96, 256, 0, stream>>>(att_b, X, c_b);

  // ---- batch 3: G[d, br] = c·ws ----
  RowDotBatch b3 = {};
  b3.j[0] = {c_f, ft_ws, nullptr, nullptr, Gm + 0,  32, 768};
  b3.j[1] = {c_f, ft_we, nullptr, nullptr, Gm + 32, 32, 768};
  b3.j[2] = {c_b, bh_ws, nullptr, nullptr, Gm + 64, 32, 768};
  b3.j[3] = {c_b, bh_we, nullptr, nullptr, Gm + 96, 32, 768};
  k_rowdot_batch<<<dim3(1, 4), 256, 0, stream>>>(b3);

  // ---- final broadcast add into ft_s/ft_e/bh_s/bh_e ----
  k_final<<<(262144 + 255) / 256, 256, 0, stream>>>(Pm, Qm, Gm, ft_bs, ft_be, bh_bs, bh_be,
                                                    out + 4096);
}
